// WindowChannelAttention_42305427866177
// MI455X (gfx1250) — matrix operand order repaired and checked by reading, not compiled, not run on hardware
//
#include <hip/hip_runtime.h>
#include <math.h>

typedef __attribute__((ext_vector_type(16))) _Float16 v16h;
typedef __attribute__((ext_vector_type(8)))  float    v8f;

__device__ __forceinline__ int lane_id() { return (int)(threadIdx.x & 31); }

// ---- DPP16 butterfly helpers: pure-VALU cross-lane moves within each 16-lane row ----
template <int CTRL>
__device__ __forceinline__ float dppf(float x) {
  int i = __builtin_bit_cast(int, x);
  int r = __builtin_amdgcn_update_dpp(i, i, CTRL, 0xf, 0xf, true);
  return __builtin_bit_cast(float, r);
}

#define BFLY_STEP(CTRL, OP, m)                                                   \
  {                                                                              \
    float t_[8];                                                                 \
    _Pragma("unroll") for (int i_ = 0; i_ < 8; ++i_) t_[i_] = dppf<CTRL>(m[i_]); \
    _Pragma("unroll") for (int i_ = 0; i_ < 8; ++i_) m[i_] = OP(m[i_], t_[i_]);  \
  }

__device__ __forceinline__ float fadd2(float a, float b) { return a + b; }

__device__ __forceinline__ void reduce_max_rows(float m[8]) {
  BFLY_STEP(0xB1, fmaxf, m) BFLY_STEP(0x4E, fmaxf, m)
  BFLY_STEP(0x141, fmaxf, m) BFLY_STEP(0x140, fmaxf, m)
}
__device__ __forceinline__ void reduce_sum_rows(float m[8]) {
  BFLY_STEP(0xB1, fadd2, m) BFLY_STEP(0x4E, fadd2, m)
  BFLY_STEP(0x141, fadd2, m) BFLY_STEP(0x140, fadd2, m)
}

// Packed f32->f16 convert + contiguous 4-half LDS store (dst must be 4B aligned).
__device__ __forceinline__ void store_pk4(_Float16* dst, float4 f) {
  auto lo = __builtin_amdgcn_cvt_pkrtz(f.x, f.y);
  auto hi = __builtin_amdgcn_cvt_pkrtz(f.z, f.w);
  *reinterpret_cast<decltype(lo)*>(dst)     = lo;
  *reinterpret_cast<decltype(hi)*>(dst + 2) = hi;
}

// A-matrix (16xK strip, f16) gather per CDNA5 wave32 layout:
// lane<16: K = k0+{0..7, 16..23}; lane>=16: K = k0+{8..15, 24..31}; row = row0 + (lane&15)
__device__ __forceinline__ v16h load_A16(const _Float16* lds, int stride, int row0, int k0) {
  const int lane = lane_id();
  const _Float16* p = lds + (row0 + (lane & 15)) * stride + k0 + ((lane >> 4) << 3);
  v16h a;
#pragma unroll
  for (int s = 0; s < 8; ++s) { a[s] = p[s]; a[s + 8] = p[s + 16]; }
  return a;
}

// B-matrix (32x16, f16) gather; source LDS laid out as [n][k]:
// lane<16: K = k0+{0..7, 16..23}; lane>=16: K = k0+{8..15, 24..31}; col = n0 + (lane&15)
__device__ __forceinline__ v16h load_B16(const _Float16* ldsT, int stride, int n0, int k0) {
  const int lane = lane_id();
  const _Float16* p = ldsT + (n0 + (lane & 15)) * stride + k0 + ((lane >> 4) << 3);
  v16h b;
#pragma unroll
  for (int s = 0; s < 8; ++s) { b[s] = p[s]; b[s + 8] = p[s + 16]; }
  return b;
}

__device__ __forceinline__ v8f wmma16(v16h a, v16h b, v8f c) {
  return __builtin_amdgcn_wmma_f32_16x16x32_f16(false, a, false, b, (short)0, c, false, false);
}

// -------------------- Window attention (self-attention of q, w=8, d=32) --------------------
#define SQ  40   // halves; 80B rows (16B aligned)
#define SQT 72   // 144B rows
#define SP  72

__global__ __launch_bounds__(128) void win_attn_kernel(const float* __restrict__ q,
                                                       float* __restrict__ out) {
  __shared__ _Float16 ldsQ [64 * SQ];   // [q][d]   (A for S; [n][k] for B=K^T)
  __shared__ _Float16 ldsQt[32 * SQT];  // [d][q]   (B source for O = P*V, V=Q)
  __shared__ _Float16 ldsP [64 * SP];   // [q][key]

  const int wb   = (int)blockIdx.x;
  const int head = wb & 7;
  const int ni   = (wb >> 3) & 15;
  const int mi   = (wb >> 7) & 15;
  const int bi   = wb >> 11;

  const int tid = (int)threadIdx.x;
  // stage Q tile: 512 float4s (vectorized along w2)
  for (int e = tid; e < 512; e += 128) {
    int c4 = e & 1, w1 = (e >> 1) & 7, d = e >> 4;
    size_t ga = ((size_t)(bi * 256 + head * 32 + d) * 128 + mi * 8 + w1) * 128 + ni * 8 + c4 * 4;
    float4 f = *reinterpret_cast<const float4*>(q + ga);
    int ql = w1 * 8 + c4 * 4;
    // scattered (per-row) writes: scalar converts
    ldsQ[(ql + 0) * SQ + d] = (_Float16)f.x;
    ldsQ[(ql + 1) * SQ + d] = (_Float16)f.y;
    ldsQ[(ql + 2) * SQ + d] = (_Float16)f.z;
    ldsQ[(ql + 3) * SQ + d] = (_Float16)f.w;
    // contiguous run: packed converts
    store_pk4(ldsQt + d * SQT + ql, f);
  }
  __syncthreads();

  const int wv   = tid >> 5;
  const int r0   = wv * 16;
  const int lane = tid & 31;
  const int hf   = lane >> 4;
  const float scale = 0.17677669529663687f; // 1/sqrt(32)

  v8f s[4];
#pragma unroll
  for (int j = 0; j < 4; ++j)
#pragma unroll
    for (int i = 0; i < 8; ++i) s[j][i] = 0.f;

  { // S = Q*Q^T : K=32 -> one k-step, 4 column tiles
    v16h a = load_A16(ldsQ, SQ, r0, 0);
#pragma unroll
    for (int j = 0; j < 4; ++j)
      s[j] = wmma16(a, load_B16(ldsQ, SQ, 16 * j, 0), s[j]);
  }

  // softmax in the raw-score domain; scale folded into exp via fma
  float mx[8], sm[8];
#pragma unroll
  for (int i = 0; i < 8; ++i) { mx[i] = -1e30f; sm[i] = 0.f; }
#pragma unroll
  for (int j = 0; j < 4; ++j)
#pragma unroll
    for (int i = 0; i < 8; ++i) mx[i] = fmaxf(mx[i], s[j][i]);
  reduce_max_rows(mx);
#pragma unroll
  for (int i = 0; i < 8; ++i) mx[i] = -mx[i] * scale; // -max*scale
#pragma unroll
  for (int j = 0; j < 4; ++j)
#pragma unroll
    for (int i = 0; i < 8; ++i) {
      float e = __expf(fmaf(s[j][i], scale, mx[i]));
      s[j][i] = e; sm[i] += e;
    }
  reduce_sum_rows(sm);
#pragma unroll
  for (int i = 0; i < 8; ++i) sm[i] = 1.f / sm[i]; // applied at O store

#pragma unroll
  for (int j = 0; j < 4; ++j)
#pragma unroll
    for (int i = 0; i < 8; ++i)
      ldsP[(r0 + i + 8 * hf) * SP + 16 * j + (lane & 15)] = (_Float16)s[j][i];
  __syncthreads();

  v8f o[2];
#pragma unroll
  for (int nt = 0; nt < 2; ++nt)
#pragma unroll
    for (int i = 0; i < 8; ++i) o[nt][i] = 0.f;
#pragma unroll
  for (int ks = 0; ks < 2; ++ks) { // O = P*Q : K=64 -> 2 k-steps, d=32 -> 2 column tiles
    v16h a = load_A16(ldsP, SP, r0, 32 * ks);
#pragma unroll
    for (int nt = 0; nt < 2; ++nt)
      o[nt] = wmma16(a, load_B16(ldsQt, SQT, 16 * nt, 32 * ks), o[nt]);
  }
#pragma unroll
  for (int nt = 0; nt < 2; ++nt)
#pragma unroll
    for (int i = 0; i < 8; ++i) {
      int ql = r0 + i + 8 * hf;
      int d  = nt * 16 + (lane & 15);
      size_t ga = ((size_t)(bi * 256 + head * 32 + d) * 128 + mi * 8 + (ql >> 3)) * 128 + ni * 8 + (ql & 7);
      out[ga] = o[nt][i] * sm[i];
    }
}

// -------------------- Channel attention (flash over 256 channels, d=128) --------------------
#define SQC 136  // 272B rows (16B aligned)
#define SKC 136
#define SVT 72
#define SPC 72

__global__ __launch_bounds__(128) void chan_attn_kernel(const float* __restrict__ q,
                                                        const float* __restrict__ k,
                                                        const float* __restrict__ v,
                                                        float* __restrict__ out) {
  __shared__ _Float16 ldsQ [64  * SQC];  // [c][d]
  __shared__ _Float16 ldsK [64  * SKC];  // [key][d] ( == [n][k] for B=K^T )
  __shared__ _Float16 ldsVt[128 * SVT];  // [d][key]
  __shared__ _Float16 ldsP [64  * SPC];  // [c][key]

  const int blk   = (int)blockIdx.x;
  const int strip = blk & 3;
  const int g     = (blk >> 2) & 127;
  const int bi    = blk >> 9;
  const int c0    = strip * 64;

  const int tid = (int)threadIdx.x;
  for (int e = tid; e < 2048; e += 128) { // 64 rows x 32 float4
    int dq = (e & 31) << 2, c = e >> 5;
    size_t ga = (size_t)(bi * 256 + c0 + c) * 16384 + g * 128 + dq;
    store_pk4(ldsQ + c * SQC + dq, *reinterpret_cast<const float4*>(q + ga));
  }

  const int wv   = tid >> 5;
  const int r0   = wv * 16;
  const int lane = tid & 31;
  const int hf   = lane >> 4;
  const float scale = 0.08838834764831845f; // 1/sqrt(128)

  v8f o[8]; float m[8], l[8];
#pragma unroll
  for (int nt = 0; nt < 8; ++nt)
#pragma unroll
    for (int i = 0; i < 8; ++i) o[nt][i] = 0.f;
#pragma unroll
  for (int i = 0; i < 8; ++i) { m[i] = -1e30f; l[i] = 0.f; }

  for (int kc = 0; kc < 4; ++kc) {
    __syncthreads();
    for (int e = tid; e < 2048; e += 128) { // 64 keys x 32 float4
      int dq = (e & 31) << 2, key = e >> 5;
      size_t ga = (size_t)(bi * 256 + kc * 64 + key) * 16384 + g * 128 + dq;
      float4 fk = *reinterpret_cast<const float4*>(k + ga);
      float4 fv = *reinterpret_cast<const float4*>(v + ga);
      store_pk4(ldsK + key * SKC + dq, fk); // contiguous: packed converts
      ldsVt[(dq + 0) * SVT + key] = (_Float16)fv.x; // transposed: scalar
      ldsVt[(dq + 1) * SVT + key] = (_Float16)fv.y;
      ldsVt[(dq + 2) * SVT + key] = (_Float16)fv.z;
      ldsVt[(dq + 3) * SVT + key] = (_Float16)fv.w;
    }
    // Warm L2 with the next chunk while this chunk's WMMAs run (global_prefetch_b8).
    if (kc < 3) {
      const float* nk = k + (size_t)(bi * 256 + (kc + 1) * 64) * 16384 + g * 128;
      const float* nv = v + (size_t)(bi * 256 + (kc + 1) * 64) * 16384 + g * 128;
      for (int e = tid; e < 256; e += 128) { // 64 rows x 4 cachelines(128B)
        int row = e >> 2, off = (e & 3) * 32;
        __builtin_prefetch(nk + (size_t)row * 16384 + off, 0, 1);
        __builtin_prefetch(nv + (size_t)row * 16384 + off, 0, 1);
      }
    }
    __syncthreads();

    v8f s[4];
#pragma unroll
    for (int j = 0; j < 4; ++j)
#pragma unroll
      for (int i = 0; i < 8; ++i) s[j][i] = 0.f;
#pragma unroll
    for (int ks = 0; ks < 4; ++ks) { // K=128 -> 4 k-steps
      v16h a = load_A16(ldsQ, SQC, r0, 32 * ks);
#pragma unroll
      for (int j = 0; j < 4; ++j)
        s[j] = wmma16(a, load_B16(ldsK, SKC, 16 * j, 32 * ks), s[j]);
    }

    // flash update in raw-score domain; scale folded via fma
    float cm[8], cs[8], nm[8], al[8];
#pragma unroll
    for (int i = 0; i < 8; ++i) { cm[i] = -1e30f; cs[i] = 0.f; }
#pragma unroll
    for (int j = 0; j < 4; ++j)
#pragma unroll
      for (int i = 0; i < 8; ++i) cm[i] = fmaxf(cm[i], s[j][i]);
    reduce_max_rows(cm);
#pragma unroll
    for (int i = 0; i < 8; ++i) {
      nm[i] = fmaxf(m[i], cm[i]);
      al[i] = __expf((m[i] - nm[i]) * scale);
      m[i]  = nm[i];
      nm[i] = -nm[i] * scale; // -newmax*scale for the fma below
    }
#pragma unroll
    for (int j = 0; j < 4; ++j)
#pragma unroll
      for (int i = 0; i < 8; ++i) {
        float e = __expf(fmaf(s[j][i], scale, nm[i]));
        s[j][i] = e; cs[i] += e;
      }
    reduce_sum_rows(cs);
#pragma unroll
    for (int i = 0; i < 8; ++i) l[i] = l[i] * al[i] + cs[i];
#pragma unroll
    for (int nt = 0; nt < 8; ++nt)
#pragma unroll
      for (int i = 0; i < 8; ++i) o[nt][i] = o[nt][i] * al[i];

#pragma unroll
    for (int j = 0; j < 4; ++j)
#pragma unroll
      for (int i = 0; i < 8; ++i)
        ldsP[(r0 + i + 8 * hf) * SPC + 16 * j + (lane & 15)] = (_Float16)s[j][i];
    __syncthreads();

#pragma unroll
    for (int ks = 0; ks < 2; ++ks) { // 64-key chunk -> 2 k-steps; d=128 -> 8 col tiles
      v16h a = load_A16(ldsP, SPC, r0, 32 * ks);
#pragma unroll
      for (int nt = 0; nt < 8; ++nt)
        o[nt] = wmma16(a, load_B16(ldsVt, SVT, 16 * nt, 32 * ks), o[nt]);
    }
  }

#pragma unroll
  for (int i = 0; i < 8; ++i) l[i] = 1.f / l[i];
#pragma unroll
  for (int nt = 0; nt < 8; ++nt)
#pragma unroll
    for (int i = 0; i < 8; ++i) {
      int c = c0 + r0 + i + 8 * hf;
      int d = nt * 16 + (lane & 15);
      size_t ga = (size_t)(bi * 256 + c) * 16384 + g * 128 + d;
      out[ga] += o[nt][i] * l[i]; // accumulate on top of window-attention result
    }
}

extern "C" void kernel_launch(void* const* d_in, const int* in_sizes, int n_in,
                              void* d_out, int out_size, void* d_ws, size_t ws_size,
                              hipStream_t stream) {
  (void)in_sizes; (void)n_in; (void)d_ws; (void)ws_size; (void)out_size;
  const float* q   = (const float*)d_in[0];
  const float* k   = (const float*)d_in[1];
  // d_in[2] = v_spatial is unused (kept source bug: window attention uses q for q/k/v)
  const float* vch = (const float*)d_in[3];
  float* out = (float*)d_out;

  win_attn_kernel<<<dim3(8192), dim3(128), 0, stream>>>(q, out);
  chan_attn_kernel<<<dim3(2048), dim3(128), 0, stream>>>(q, k, vch, out);
}
